// MultiHeadAttention_29884382445713
// MI455X (gfx1250) — compile-verified, matrix operand order checked by reading, not run on hardware
//
#include <hip/hip_runtime.h>
#include <hip/hip_bf16.h>
#include <stdint.h>

// MHA forward for gfx1250 (MI455X). All GEMMs via v_wmma_f32_16x16x32_f16.
// Flash-attention structure (S x S scores never materialized). K/V tiles
// staged in LDS by the Tensor Data Mover (double-buffered, TENSORcnt-gated).
// GEMM kernels use the low-pressure shape (A frag reused 12x, B frags loaded
// inline) that the compiler fully unrolls without spilling; memory latency is
// hidden by 8-wave occupancy, not in-wave double buffering (which spilled).

#define B_ 8
#define S_ 1024
#define D_ 1024
#define H_ 16
#define DH_ 64
#define SCALE_ 0.03125f /* 1/sqrt(1024) */

typedef __attribute__((ext_vector_type(16))) _Float16 v16h;
typedef __attribute__((ext_vector_type(8)))  float    v8f;
typedef __attribute__((ext_vector_type(4)))  unsigned int v4u;
typedef __attribute__((ext_vector_type(4)))  unsigned int u32x4;
typedef __attribute__((ext_vector_type(8)))  int      i32x8;
typedef __attribute__((ext_vector_type(4)))  int      i32x4;

union Frag16 { v16h h; v4u u[2]; };

// 16x32 f16 A/B fragment (ISA 7.12.2): lane element e holds
// k = kbase + (e%8) + 16*(e/8) + 8*(lane/16) -> two 16B loads at +0 / +16 halves.
static __device__ __forceinline__ v16h load_row_frag(const _Float16* __restrict__ row,
                                                     int kbase, int lh) {
  Frag16 f;
  const v4u* p = (const v4u*)(row + kbase + 8 * lh);
  f.u[0] = p[0];
  f.u[1] = p[2];
  return f.h;
}

static __device__ __forceinline__ v8f wmma16(v16h a, v16h b, v8f c) {
  return __builtin_amdgcn_wmma_f32_16x16x32_f16(false, a, false, b, (short)0, c,
                                                false, false);
}

static __device__ __forceinline__ float rowmax16(float v) {
#pragma unroll
  for (int m = 1; m < 16; m <<= 1) v = fmaxf(v, __shfl_xor(v, m, 32));
  return v;
}
static __device__ __forceinline__ float rowsum16(float v) {
#pragma unroll
  for (int m = 1; m < 16; m <<= 1) v += __shfl_xor(v, m, 32);
  return v;
}

// TDM 2D tile load: D# per ISA §8.3/8.4. data_size=2B, type=2("image"),
// groups 2/3 zero (2D tile). dims/strides in elements, lds_off in bytes.
// This toolchain exposes the 6-arg builtin (extra int32x8, zero-filled).
static __device__ __forceinline__ void tdm_load_2d(unsigned lds_off, const void* gptr,
                                                   unsigned dim0, unsigned dim1,
                                                   unsigned stride0,
                                                   unsigned tile0, unsigned tile1) {
  unsigned long long ga = (unsigned long long)(uintptr_t)gptr;
  u32x4 g0;
  g0[0] = 1u;                                            // count=1, user descriptor
  g0[1] = lds_off;                                       // lds_addr (bytes)
  g0[2] = (unsigned)(ga & 0xFFFFFFFFu);                  // global_addr[31:0]
  g0[3] = (unsigned)((ga >> 32) & 0x01FFFFFFu) | 0x80000000u; // addr[56:32] | type=2
  i32x8 g1;
  g1[0] = (int)(1u << 16);                               // data_size=1 -> 2 bytes
  g1[1] = (int)(dim0 << 16);                             // tensor_dim0[15:0]
  g1[2] = (int)((dim0 >> 16) | (dim1 << 16));            // dim0[31:16] | dim1[15:0]
  g1[3] = (int)((dim1 >> 16) | (tile0 << 16));           // dim1[31:16] | tile_dim0
  g1[4] = (int)(tile1 & 0xFFFFu);                        // tile_dim1 (tile_dim2=0)
  g1[5] = (int)stride0;                                  // tensor_dim0_stride[31:0]
  g1[6] = 0;
  g1[7] = 0;
  i32x4 z4 = {0, 0, 0, 0};
  i32x8 z8 = {0, 0, 0, 0, 0, 0, 0, 0};
  __builtin_amdgcn_tensor_load_to_lds(g0, g1, z4, z4, z8, 0);
}

// ---------------- prep kernels ----------------
__global__ void cvt_x_kernel(const float* __restrict__ x, _Float16* __restrict__ x16, int n) {
  int i = blockIdx.x * blockDim.x + threadIdx.x;
  if (i < n) x16[i] = (_Float16)x[i];
}

// wT[mat][h][e][d] = W{q,k,v}[h][d][e]  (f16)
__global__ void tr_qkv_kernel(const float* __restrict__ Wq, const float* __restrict__ Wk,
                              const float* __restrict__ Wv, _Float16* __restrict__ wT) {
  int i = blockIdx.x * blockDim.x + threadIdx.x;
  const int total = 3 * H_ * DH_ * D_;
  if (i >= total) return;
  int d = i % D_;
  int r = i / D_;
  int e = r % DH_; r /= DH_;
  int h = r % H_;
  int mat = r / H_;
  const float* W = (mat == 0) ? Wq : ((mat == 1) ? Wk : Wv);
  wT[i] = (_Float16)W[((size_t)h * D_ + d) * DH_ + e];
}

// woT[n][k] = Wo[k][n]  (f16)
__global__ void tr_wo_kernel(const float* __restrict__ Wo, _Float16* __restrict__ woT) {
  int i = blockIdx.x * blockDim.x + threadIdx.x;
  if (i >= D_ * D_) return;
  int k = i % D_;
  int n = i / D_;
  woT[(size_t)n * D_ + k] = (_Float16)Wo[(size_t)k * D_ + n];
}

// ---------------- QKV projection ----------------
// One wave per 16-row strip of one (b,h): 12 accumulators (Q,K,V x 4 n-tiles),
// A fragment (X tile) loaded once per k-step and reused 12x. B fragments
// loaded inline (low pressure -> compiler fully unrolls, no spills).
__global__ void qkv_kernel(const _Float16* __restrict__ x16, const _Float16* __restrict__ wT,
                           const float* __restrict__ bq, const float* __restrict__ bk,
                           const float* __restrict__ bv,
                           _Float16* __restrict__ Q, _Float16* __restrict__ K,
                           _Float16* __restrict__ Vt) {
  const int w  = blockIdx.x * (blockDim.x >> 5) + (threadIdx.x >> 5);
  const int mt = w & 63;        // S/16 strips
  const int bh = w >> 6;        // b*H + h
  const int b  = bh >> 4;
  const int h  = bh & 15;
  const int lane = threadIdx.x & 31;
  const int lm = lane & 15;
  const int lh = lane >> 4;

  const _Float16* arow = x16 + ((size_t)b * S_ + mt * 16 + lm) * D_;

  const _Float16* brow[3][4];
#pragma unroll
  for (int mat = 0; mat < 3; ++mat)
#pragma unroll
    for (int nt = 0; nt < 4; ++nt)
      brow[mat][nt] = wT + (((size_t)mat * H_ + h) * DH_ + nt * 16 + lm) * D_;

  v8f acc[3][4];
#pragma unroll
  for (int mat = 0; mat < 3; ++mat)
#pragma unroll
    for (int nt = 0; nt < 4; ++nt) acc[mat][nt] = (v8f){};

  for (int k0 = 0; k0 < D_; k0 += 32) {
    v16h a = load_row_frag(arow, k0, lh);
#pragma unroll
    for (int mat = 0; mat < 3; ++mat)
#pragma unroll
      for (int nt = 0; nt < 4; ++nt)
        acc[mat][nt] = wmma16(a, load_row_frag(brow[mat][nt], k0, lh), acc[mat][nt]);
  }

  const float* bias[3] = {bq, bk, bv};
#pragma unroll
  for (int mat = 0; mat < 3; ++mat) {
#pragma unroll
    for (int nt = 0; nt < 4; ++nt) {
      const int col = nt * 16 + lm;
      const float bs = bias[mat][h * DH_ + col];
#pragma unroll
      for (int r = 0; r < 8; ++r) {
        const int sr = mt * 16 + r + 8 * lh;       // C/D frag row mapping
        const _Float16 v = (_Float16)(acc[mat][nt][r] + bs);
        if (mat == 0)      Q [((size_t)bh * S_ + sr) * DH_ + col] = v;
        else if (mat == 1) K [((size_t)bh * S_ + sr) * DH_ + col] = v;
        else               Vt[((size_t)bh * DH_ + col) * S_ + sr] = v; // transposed
      }
    }
  }
}

// ---------------- flash attention ----------------
// Block = 8 waves, ALL on the same (b,h), 128 query rows. K (32x64) and Vt
// (64x32) tiles staged in LDS by the TDM, double-buffered; wave 0 issues the
// next tile's descriptors then s_wait_tensorcnt(2) covers the current tile
// (TDM completes in order per wave). K/V global traffic cut 8x.
__global__ void attn_kernel(const _Float16* __restrict__ Q, const _Float16* __restrict__ K,
                            const _Float16* __restrict__ Vt,
                            const unsigned char* __restrict__ mask,
                            _Float16* __restrict__ ctx16) {
  __shared__ _Float16 kbuf[2][32 * 64];   // rows j0..j0+31, 64 halves each
  __shared__ _Float16 vbuf[2][64 * 32];   // Vt rows 0..63, 32 halves each
  __shared__ _Float16 pbuf[8][16 * 32];   // per-wave P tile

  const int widx = threadIdx.x >> 5;
  const int bh = blockIdx.x >> 3;          // 8 blocks per (b,h)
  const int mg = blockIdx.x & 7;
  const int mt = mg * 8 + widx;            // this wave's 16-row strip
  const int b  = bh >> 4;
  const int h  = bh & 15;
  const int lane = threadIdx.x & 31;
  const int lm = lane & 15;
  const int lh = lane >> 4;

  _Float16* pb = pbuf[widx];
  const _Float16* Kbh  = K  + (size_t)bh * S_ * DH_;
  const _Float16* Vtbh = Vt + (size_t)bh * DH_ * S_;

  const _Float16* qrow = Q + ((size_t)bh * S_ + mt * 16 + lm) * DH_;
  const v16h qf0 = load_row_frag(qrow, 0, lh);
  const v16h qf1 = load_row_frag(qrow, 32, lh);

  v8f acc[4];
#pragma unroll
  for (int n = 0; n < 4; ++n) acc[n] = (v8f){};
  float mrun[8], lrun[8];
#pragma unroll
  for (int r = 0; r < 8; ++r) { mrun[r] = -INFINITY; lrun[r] = 0.0f; }

  // preload tile 0
  if (widx == 0) {
    tdm_load_2d((unsigned)(uintptr_t)&kbuf[0][0], Kbh,  DH_, S_, DH_, DH_, 32);
    tdm_load_2d((unsigned)(uintptr_t)&vbuf[0][0], Vtbh, S_, DH_, S_,  32, DH_);
  }

  int stage = 0;
  for (int j0 = 0; j0 < S_; j0 += 32) {
    if (widx == 0) {
      if (j0 + 32 < S_) {
        tdm_load_2d((unsigned)(uintptr_t)&kbuf[stage ^ 1][0], Kbh + (size_t)(j0 + 32) * DH_,
                    DH_, S_, DH_, DH_, 32);
        tdm_load_2d((unsigned)(uintptr_t)&vbuf[stage ^ 1][0], Vtbh + (j0 + 32),
                    S_, DH_, S_, 32, DH_);
        __builtin_amdgcn_s_wait_tensorcnt(2);  // current tile's 2 loads done
      } else {
        __builtin_amdgcn_s_wait_tensorcnt(0);
      }
    }
    __syncthreads();   // publish kbuf/vbuf[stage]

    const _Float16* kb = kbuf[stage];
    const _Float16* vb = vbuf[stage];

    // two 16x16 score tiles from LDS-staged K
    v8f sc0 = (v8f){}, sc1 = (v8f){};
    {
      const _Float16* kr0 = kb + (size_t)lm * DH_;
      sc0 = wmma16(qf0, load_row_frag(kr0, 0, lh), sc0);
      sc0 = wmma16(qf1, load_row_frag(kr0, 32, lh), sc0);
      const _Float16* kr1 = kb + (size_t)(16 + lm) * DH_;
      sc1 = wmma16(qf0, load_row_frag(kr1, 0, lh), sc1);
      sc1 = wmma16(qf1, load_row_frag(kr1, 32, lh), sc1);
    }

    // scale + mask + online softmax
    float p0a[8], p1a[8], alpha[8];
#pragma unroll
    for (int r = 0; r < 8; ++r) {
      const int qr = mt * 16 + r + 8 * lh;
      const unsigned char* mrow = mask + ((size_t)b * S_ + qr) * S_ + j0;
      float s0 = sc0[r] * SCALE_; if (mrow[lm])      s0 = -1.0e9f;
      float s1 = sc1[r] * SCALE_; if (mrow[16 + lm]) s1 = -1.0e9f;
      const float tmax = rowmax16(fmaxf(s0, s1));
      const float mn = fmaxf(mrun[r], tmax);
      alpha[r] = __expf(mrun[r] - mn);
      const float p0 = __expf(s0 - mn);
      const float p1 = __expf(s1 - mn);
      const float rs = rowsum16(p0 + p1);
      lrun[r] = lrun[r] * alpha[r] + rs;
      mrun[r] = mn;
      p0a[r] = p0; p1a[r] = p1;
    }
#pragma unroll
    for (int n = 0; n < 4; ++n)
#pragma unroll
      for (int r = 0; r < 8; ++r) acc[n][r] *= alpha[r];

    // P (f16) -> wave-private LDS tile, then re-gather as A fragment
#pragma unroll
    for (int r = 0; r < 8; ++r) {
      const int row = r + 8 * lh;
      pb[row * 32 + lm]      = (_Float16)p0a[r];
      pb[row * 32 + 16 + lm] = (_Float16)p1a[r];
    }
    asm volatile("s_wait_dscnt 0" ::: "memory");
    Frag16 pa;
    {
      const v4u* q = (const v4u*)(pb + lm * 32 + 8 * lh);
      pa.u[0] = q[0];
      pa.u[1] = q[2];
    }

    // ctx += P(16x32) x V(32x64) from LDS-staged Vt
#pragma unroll
    for (int n = 0; n < 4; ++n) {
      const _Float16* vr = vb + (size_t)(n * 16 + lm) * 32;
      acc[n] = wmma16(pa.h, load_row_frag(vr, 0, lh), acc[n]);
    }

    __syncthreads();   // all waves done reading stage before TDM overwrites it
    stage ^= 1;
  }

  // normalize, store ctx in concat-head layout [B, S, H*DH] (f16)
#pragma unroll
  for (int n = 0; n < 4; ++n) {
    const int col = h * DH_ + n * 16 + lm;
#pragma unroll
    for (int r = 0; r < 8; ++r) {
      const int sr = mt * 16 + r + 8 * lh;
      ctx16[((size_t)b * S_ + sr) * D_ + col] = (_Float16)(acc[n][r] / lrun[r]);
    }
  }
}

// ---------------- output projection ----------------
__global__ void proj_kernel(const _Float16* __restrict__ ctx16, const _Float16* __restrict__ woT,
                            const float* __restrict__ bo, float* __restrict__ out) {
  const int w  = blockIdx.x * (blockDim.x >> 5) + (threadIdx.x >> 5);
  const int ng = w & 15;        // 64-col group
  const int mt = w >> 4;        // 16-row strip over B*S
  const int lane = threadIdx.x & 31;
  const int lm = lane & 15;
  const int lh = lane >> 4;

  const _Float16* arow = ctx16 + (size_t)(mt * 16 + lm) * D_;
  const _Float16* brow[4];
#pragma unroll
  for (int n = 0; n < 4; ++n) brow[n] = woT + (size_t)(ng * 64 + n * 16 + lm) * D_;

  v8f acc[4];
#pragma unroll
  for (int n = 0; n < 4; ++n) acc[n] = (v8f){};

  for (int k0 = 0; k0 < D_; k0 += 32) {
    v16h a = load_row_frag(arow, k0, lh);
#pragma unroll
    for (int n = 0; n < 4; ++n)
      acc[n] = wmma16(a, load_row_frag(brow[n], k0, lh), acc[n]);
  }

#pragma unroll
  for (int n = 0; n < 4; ++n) {
    const int col = ng * 64 + n * 16 + lm;
    const float bs = bo[col];
#pragma unroll
    for (int r = 0; r < 8; ++r) {
      const int row = mt * 16 + r + 8 * lh;
      out[(size_t)row * D_ + col] = acc[n][r] + bs;
    }
  }
}

extern "C" void kernel_launch(void* const* d_in, const int* in_sizes, int n_in,
                              void* d_out, int out_size, void* d_ws, size_t ws_size,
                              hipStream_t stream) {
  const float* x  = (const float*)d_in[0];
  const unsigned char* mask = (const unsigned char*)d_in[1]; // bool [B,S,S]
  const float* Wq = (const float*)d_in[2];
  const float* bq = (const float*)d_in[3];
  const float* Wk = (const float*)d_in[4];
  const float* bk = (const float*)d_in[5];
  const float* Wv = (const float*)d_in[6];
  const float* bv = (const float*)d_in[7];
  const float* Wo = (const float*)d_in[8];
  const float* bo = (const float*)d_in[9];
  float* out = (float*)d_out;

  // workspace carve-up (~92.3 MB)
  char* ws = (char*)d_ws;
  const size_t nX   = (size_t)B_ * S_ * D_;
  const size_t nWT  = (size_t)3 * H_ * DH_ * D_;
  const size_t nWoT = (size_t)D_ * D_;
  const size_t nQ   = (size_t)B_ * H_ * S_ * DH_;
  _Float16* x16 = (_Float16*)ws;  ws += nX   * 2;
  _Float16* wT  = (_Float16*)ws;  ws += nWT  * 2;
  _Float16* woT = (_Float16*)ws;  ws += nWoT * 2;
  _Float16* Qb  = (_Float16*)ws;  ws += nQ   * 2;
  _Float16* Kb  = (_Float16*)ws;  ws += nQ   * 2;
  _Float16* Vt  = (_Float16*)ws;  ws += nQ   * 2;
  _Float16* ctx = (_Float16*)ws;  ws += nX   * 2;

  cvt_x_kernel<<<(int)((nX + 255) / 256), 256, 0, stream>>>(x, x16, (int)nX);
  tr_qkv_kernel<<<(int)((nWT + 255) / 256), 256, 0, stream>>>(Wq, Wk, Wv, wT);
  tr_wo_kernel<<<(int)((nWoT + 255) / 256), 256, 0, stream>>>(Wo, woT);

  // B*H*(S/16) = 8192 waves, 8 per block
  qkv_kernel<<<1024, 256, 0, stream>>>(x16, wT, bq, bk, bv, Qb, Kb, Vt);
  // B*H*(S/128) = 1024 blocks, each 8 waves on one (b,h)
  attn_kernel<<<1024, 256, 0, stream>>>(Qb, Kb, Vt, mask, ctx);
  // (B*S/16)*(D/64) = 8192 waves
  proj_kernel<<<1024, 256, 0, stream>>>(ctx, woT, bo, out);
}
